// GNS_43370579755173
// MI455X (gfx1250) — matrix operand order repaired and checked by reading, not compiled, not run on hardware
//
#include <hip/hip_runtime.h>

typedef __attribute__((ext_vector_type(16))) _Float16 v16h;
typedef __attribute__((ext_vector_type(8)))  float    v8f;
typedef __attribute__((ext_vector_type(4)))  unsigned int u32x4;
typedef __attribute__((ext_vector_type(4)))  int i32x4;
typedef __attribute__((ext_vector_type(8)))  int i32x8;

#define THREADS 256
#define XP  136            // LDS activation row stride (halfs): 272B, 16B-aligned, 4-bank rotate
#define KPW 136            // LDS weight col stride (halfs)
#define BLKH (128 * KPW)   // one weight block: 17408 halfs = 34816 B
#define OFFW0 ((unsigned)(128 * XP * 2))            // LDS byte offset of weight buffer 0
#define OFFW1 ((unsigned)(128 * XP * 2 + BLKH * 2)) // LDS byte offset of weight buffer 1
#define SMEM_HALFS (128 * XP + 2 * BLKH)            // 104448 B total
#define NCPY8 ((unsigned)(BLKH * 2 / 8))            // 4352 x 8B per block

#if __has_builtin(__builtin_amdgcn_tensor_load_to_lds) && __has_builtin(__builtin_amdgcn_s_wait_tensorcnt)
#define USE_TDM 1
#else
#define USE_TDM 0
#endif

union ABFrag { v16h v; u32x4 q[2]; };

static __device__ __forceinline__ v8f vzero() {
  v8f z = {0.f, 0.f, 0.f, 0.f, 0.f, 0.f, 0.f, 0.f};
  return z;
}

static __device__ __forceinline__ v16h ld16(const _Float16* p0, const _Float16* p1) {
  ABFrag t;
  t.q[0] = *(const u32x4*)p0;
  t.q[1] = *(const u32x4*)p1;
  return t.v;
}

// A fragment (16x32 f16): rows m0..m0+15 of row-major X (stride xp halfs), K-slice k0
static __device__ __forceinline__ v16h a_frag_lds(const _Float16* X, int xp, int m0, int k0, int lane) {
  const _Float16* p = X + (m0 + (lane & 15)) * xp + k0 + ((lane >> 4) << 3);
  return ld16(p, p + 16);
}

// A fragment from per-lane f16 row pointer (row-major, 16B-aligned at k0)
static __device__ __forceinline__ v16h a_frag_row(const _Float16* row, int k0, int lane) {
  const _Float16* p = row + k0 + ((lane >> 4) << 3);
  return ld16(p, p + 16);
}

// A fragment from per-lane f32 row pointer (converted on the fly)
static __device__ __forceinline__ v16h a_frag_row_f32(const float* row, int k0, int lane) {
  const float* p = row + k0 + ((lane >> 4) << 3);
  v16h t;
#pragma unroll
  for (int j = 0; j < 8; ++j) t[j] = (_Float16)p[j];
#pragma unroll
  for (int j = 0; j < 8; ++j) t[8 + j] = (_Float16)p[16 + j];
  return t;
}

// B fragment (32x16 f16): cols n0..n0+15 from column-major Wc (stride KPW halfs)
static __device__ __forceinline__ v16h b_frag_lds(const _Float16* Wc, int k0, int n0, int lane) {
  const _Float16* p = Wc + (n0 + (lane & 15)) * KPW + k0 + ((lane >> 4) << 3);
  return ld16(p, p + 16);
}

static __device__ __forceinline__ v8f wmma16(v16h a, v16h b, v8f c) {
  return __builtin_amdgcn_wmma_f32_16x16x32_f16(false, a, false, b, (short)0, c, false, false);
}

// ---- weight-block staging: TDM async copy (preferred) or cooperative copy ----
#if USE_TDM
static __device__ __forceinline__ void tdm_issue(u32x4 g0, i32x8 g1) {
  i32x4 z4 = {0, 0, 0, 0};
#if defined(__clang_major__) && __clang_major__ >= 23
  i32x8 z8 = {0, 0, 0, 0, 0, 0, 0, 0};
  __builtin_amdgcn_tensor_load_to_lds(g0, g1, z4, z4, z8, 0);
#else
  __builtin_amdgcn_tensor_load_to_lds(g0, g1, z4, z4, 0);
#endif
}
#endif

// issue async fetch of one 34816B weight block into LDS (does not wait)
static __device__ __forceinline__ void fetch_w(const _Float16* gsrc, _Float16* ldst,
                                               unsigned ldsOff, int tid) {
#if USE_TDM
  (void)ldst;
  if (tid < 32) {  // one wave issues the DMA (TDM ignores EXEC; wave-uniform branch)
    unsigned long long ga = (unsigned long long)gsrc;
    u32x4 g0;
    g0[0] = 1u;                                    // count=1, user descriptor
    g0[1] = ldsOff;                                // lds_addr (bytes)
    g0[2] = (unsigned)ga;                          // global_addr[31:0]
    g0[3] = (unsigned)((ga >> 32) & 0x01FFFFFFull) | (2u << 30);  // addr[56:32] | type=2
    i32x8 g1;
    g1[0] = (int)(3u << 16);                       // data_size = 8B
    g1[1] = (int)((NCPY8 & 0xFFFFu) << 16);        // tensor_dim0[15:0]
    g1[2] = (int)((NCPY8 >> 16) | (1u << 16));     // tensor_dim0[31:16] | tensor_dim1 = 1
    g1[3] = (int)((NCPY8 & 0xFFFFu) << 16);        // tile_dim0 = NCPY8
    g1[4] = 1;                                     // tile_dim1 = 1
    g1[5] = (int)NCPY8;                            // tensor_dim0_stride
    g1[6] = 0;
    g1[7] = 0;
    tdm_issue(g0, g1);
  }
#else
  (void)ldsOff;
  for (int i = tid; i < BLKH / 8; i += THREADS)
    ((u32x4*)ldst)[i] = ((const u32x4*)gsrc)[i];
#endif
}

static __device__ __forceinline__ void fetch_wait(int tid) {
#if USE_TDM
  if (tid < 32) __builtin_amdgcn_s_wait_tensorcnt((short)0);
#else
  (void)tid;
#endif
}

static __device__ __forceinline__ void bias_relu_store(_Float16* X, const float* __restrict__ bias,
                                                       int m0, int lane, const v8f acc[8]) {
  int half8 = (lane >> 4) << 3;
  int nlo = lane & 15;
#pragma unroll
  for (int nt = 0; nt < 8; ++nt) {
    int n = nt * 16 + nlo;
    float bv = bias[n];
#pragma unroll
    for (int r = 0; r < 8; ++r) {
      float y = acc[nt][r] + bv;
      y = y > 0.f ? y : 0.f;
      X[(m0 + r + half8) * XP + n] = (_Float16)y;
    }
  }
}

static __device__ __forceinline__ void add_bias(v8f acc[8], const float* __restrict__ bias, int lane) {
  int nlo = lane & 15;
#pragma unroll
  for (int nt = 0; nt < 8; ++nt) {
    float bv = bias[nt * 16 + nlo];
#pragma unroll
    for (int r = 0; r < 8; ++r) acc[nt][r] += bv;
  }
}

// per-row LayerNorm stats via shfl-xor over the 16-lane half that owns the row
static __device__ __forceinline__ void ln_stats(const v8f acc[8], float mu[8], float rs[8]) {
#pragma unroll
  for (int r = 0; r < 8; ++r) {
    float s = 0.f;
#pragma unroll
    for (int nt = 0; nt < 8; ++nt) s += acc[nt][r];
    s += __shfl_xor(s, 1); s += __shfl_xor(s, 2);
    s += __shfl_xor(s, 4); s += __shfl_xor(s, 8);
    float m = s * (1.0f / 128.0f);
    float v = 0.f;
#pragma unroll
    for (int nt = 0; nt < 8; ++nt) { float d = acc[nt][r] - m; v += d * d; }
    v += __shfl_xor(v, 1); v += __shfl_xor(v, 2);
    v += __shfl_xor(v, 4); v += __shfl_xor(v, 8);
    mu[r] = m;
    rs[r] = rsqrtf(v * (1.0f / 128.0f) + 1e-5f);
  }
}

static __device__ __forceinline__ void gemm128_lds(const _Float16* X, const _Float16* Wc,
                                                   int m0, int lane, v8f acc[8]) {
#pragma unroll
  for (int k0 = 0; k0 < 128; k0 += 32) {
    v16h a = a_frag_lds(X, XP, m0, k0, lane);
#pragma unroll
    for (int nt = 0; nt < 8; ++nt)
      acc[nt] = wmma16(a, b_frag_lds(Wc, k0, nt * 16, lane), acc[nt]);
  }
}

// ------------------------------------------------------------- weight prep --
// fp32 row-major W[K x Nout] slice -> f16 column-major padded block [128][KPW]
__global__ __launch_bounds__(THREADS) void gns_prep_kernel(
    const float* __restrict__ src, long srcStepStride, int Ktot, int Nout,
    int nChunks, _Float16* __restrict__ dst, int stepBlocks) {
  int b = blockIdx.x;
  int s = b / nChunks, c = b - s * nChunks;
  const float* sp = src + (size_t)s * (size_t)srcStepStride;
  _Float16* dp = dst + ((size_t)s * stepBlocks + c) * (size_t)BLKH;
  for (int i = threadIdx.x; i < BLKH; i += THREADS) {
    int n = i / KPW, k = i - n * KPW;
    int kk = c * 128 + k;
    float w = (k < 128 && kk < Ktot && n < Nout) ? sp[(size_t)kk * Nout + n] : 0.f;
    dp[i] = (_Float16)w;
  }
}

// ---------------------------------------------------------------- encoder ----
__global__ __launch_bounds__(THREADS) void gns_encoder_kernel(
    const float* __restrict__ i0, int w0, const float* __restrict__ i1, int w1,
    const float* __restrict__ i2, int w2, const float* __restrict__ i3, int w3,
    int Ktot,
    const _Float16* __restrict__ wblk,
    const float* __restrict__ b0, const float* __restrict__ b1, const float* __restrict__ b2,
    const float* __restrict__ lnS, const float* __restrict__ lnO,
    float* __restrict__ out_f, _Float16* __restrict__ out_h, int rows) {
  extern __shared__ _Float16 smem[];
  _Float16* sX = smem;
  _Float16* sW0 = smem + 128 * XP;
  _Float16* sW1 = sW0 + BLKH;
  int tid = threadIdx.x, lane = tid & 31, m0 = (tid >> 5) * 16;
  int tile = blockIdx.x * 128;
  int arow = tile + m0 + (lane & 15);
  if (arow >= rows) arow = rows - 1;

  fetch_w(wblk, sW0, OFFW0, tid);
  fetch_wait(tid); __syncthreads();

  // layer 1 (K padded to 32 with zeros in both A and prestaged W)
  fetch_w(wblk + BLKH, sW1, OFFW1, tid);
  v16h a1;
  {
    int kh = (lane >> 4) << 3;
#pragma unroll
    for (int j = 0; j < 16; ++j) {
      int col = kh + (j & 7) + ((j >> 3) << 4);
      float v = 0.f;
      if (col < w0) v = i0[arow * w0 + col];
      else if (col < w0 + w1) v = i1[arow * w1 + col - w0];
      else if (col < w0 + w1 + w2) v = i2[arow * w2 + col - w0 - w1];
      else if (col < Ktot) v = i3[arow * w3 + col - w0 - w1 - w2];
      a1[j] = (_Float16)v;
    }
  }
  v8f acc[8];
#pragma unroll
  for (int nt = 0; nt < 8; ++nt)
    acc[nt] = wmma16(a1, b_frag_lds(sW0, 0, nt * 16, lane), vzero());
  fetch_wait(tid); __syncthreads();
  bias_relu_store(sX, b0, m0, lane, acc);

  // layer 2
  fetch_w(wblk + 2 * BLKH, sW0, OFFW0, tid);
#pragma unroll
  for (int nt = 0; nt < 8; ++nt) acc[nt] = vzero();
  gemm128_lds(sX, sW1, m0, lane, acc);
  fetch_wait(tid); __syncthreads();
  bias_relu_store(sX, b1, m0, lane, acc);

  // layer 3 + LayerNorm
#pragma unroll
  for (int nt = 0; nt < 8; ++nt) acc[nt] = vzero();
  gemm128_lds(sX, sW0, m0, lane, acc);
  add_bias(acc, b2, lane);
  float mu[8], rs[8];
  ln_stats(acc, mu, rs);
  int half8 = (lane >> 4) << 3, nlo = lane & 15;
  int rowBase = tile + m0 + half8;
#pragma unroll
  for (int nt = 0; nt < 8; ++nt) {
    int n = nt * 16 + nlo;
    float sc = lnS[n], of = lnO[n];
#pragma unroll
    for (int r = 0; r < 8; ++r) {
      int g = rowBase + r;
      if (g < rows) {
        float y = (acc[nt][r] - mu[r]) * rs[r] * sc + of;
        out_f[g * 128 + n] = y;
        out_h[g * 128 + n] = (_Float16)y;
      }
    }
  }
}

// ------------------------------------------------------------- edge MLP -----
__global__ __launch_bounds__(THREADS) void gns_edge_kernel(
    const _Float16* __restrict__ nodes_h,
    _Float16* __restrict__ edges_h,
    float* __restrict__ edges_f,
    float* __restrict__ aggr,
    const int* __restrict__ senders,
    const int* __restrict__ receivers,
    const _Float16* __restrict__ wblk,   // [W0c0,W0c1,W0c2,W1,W2]
    const float* __restrict__ b0, const float* __restrict__ b1, const float* __restrict__ b2,
    const float* __restrict__ lnS, const float* __restrict__ lnO,
    int E) {
  extern __shared__ _Float16 smem[];
  _Float16* sX = smem;
  _Float16* sW0 = smem + 128 * XP;
  _Float16* sW1 = sW0 + BLKH;
  int tid = threadIdx.x, lane = tid & 31, m0 = (tid >> 5) * 16;
  int tile = blockIdx.x * 128;
  int arow = tile + m0 + (lane & 15);
  if (arow >= E) arow = E - 1;
  const _Float16* src0 = nodes_h + (size_t)senders[arow] * 128;
  const _Float16* src1 = nodes_h + (size_t)receivers[arow] * 128;
  const _Float16* src2 = edges_h + (size_t)arow * 128;

  fetch_w(wblk, sW0, OFFW0, tid);
  fetch_wait(tid); __syncthreads();

  v8f acc[8];
#pragma unroll
  for (int nt = 0; nt < 8; ++nt) acc[nt] = vzero();

  // layer 1: K=384 as 3 chunks; overlap next-chunk DMA with current WMMAs
#pragma unroll
  for (int c = 0; c < 3; ++c) {
    _Float16* wc = (c & 1) ? sW1 : sW0;
    _Float16* wn = (c & 1) ? sW0 : sW1;
    unsigned wno = (c & 1) ? OFFW0 : OFFW1;
    fetch_w(wblk + (size_t)(c + 1) * BLKH, wn, wno, tid);
    const _Float16* src = (c == 0) ? src0 : (c == 1) ? src1 : src2;
#pragma unroll
    for (int k0 = 0; k0 < 128; k0 += 32) {
      v16h a = a_frag_row(src, k0, lane);
#pragma unroll
      for (int nt = 0; nt < 8; ++nt)
        acc[nt] = wmma16(a, b_frag_lds(wc, k0, nt * 16, lane), acc[nt]);
    }
    fetch_wait(tid); __syncthreads();
  }
  bias_relu_store(sX, b0, m0, lane, acc);

  // layer 2 (W1 in sW1), prefetch W2 -> sW0
  fetch_w(wblk + (size_t)4 * BLKH, sW0, OFFW0, tid);
#pragma unroll
  for (int nt = 0; nt < 8; ++nt) acc[nt] = vzero();
  gemm128_lds(sX, sW1, m0, lane, acc);
  fetch_wait(tid); __syncthreads();
  bias_relu_store(sX, b1, m0, lane, acc);

  // layer 3
#pragma unroll
  for (int nt = 0; nt < 8; ++nt) acc[nt] = vzero();
  gemm128_lds(sX, sW0, m0, lane, acc);

  add_bias(acc, b2, lane);
  float mu[8], rs[8];
  ln_stats(acc, mu, rs);
  int half8 = (lane >> 4) << 3, nlo = lane & 15;
  int rowBase = tile + m0 + half8;
  int rcv[8];
#pragma unroll
  for (int r = 0; r < 8; ++r) {
    int g = rowBase + r;
    if (g >= E) g = E - 1;
    rcv[r] = receivers[g];
  }
#pragma unroll
  for (int nt = 0; nt < 8; ++nt) {
    int n = nt * 16 + nlo;
    float sc = lnS[n], of = lnO[n];
#pragma unroll
    for (int r = 0; r < 8; ++r) {
      int g = rowBase + r;
      if (g < E) {
        float y = (acc[nt][r] - mu[r]) * rs[r] * sc + of;  // e_new
        float ev = edges_f[g * 128 + n] + y;               // residual
        edges_f[g * 128 + n] = ev;
        edges_h[g * 128 + n] = (_Float16)ev;
        atomicAdd(&aggr[rcv[r] * 128 + n], y);             // segment_sum
      }
    }
  }
}

// ------------------------------------------------------------- node MLP -----
__global__ __launch_bounds__(THREADS) void gns_node_kernel(
    _Float16* __restrict__ nodes_h,
    float* __restrict__ nodes_f,
    const float* __restrict__ aggr,
    const _Float16* __restrict__ wblk,   // [W0c0,W0c1,W1,W2]
    const float* __restrict__ b0, const float* __restrict__ b1, const float* __restrict__ b2,
    const float* __restrict__ lnS, const float* __restrict__ lnO,
    int Nn) {
  extern __shared__ _Float16 smem[];
  _Float16* sX = smem;
  _Float16* sW0 = smem + 128 * XP;
  _Float16* sW1 = sW0 + BLKH;
  int tid = threadIdx.x, lane = tid & 31, m0 = (tid >> 5) * 16;
  int tile = blockIdx.x * 128;
  int arow = tile + m0 + (lane & 15);
  if (arow >= Nn) arow = Nn - 1;
  const _Float16* pN = nodes_h + (size_t)arow * 128;
  const float*    pA = aggr    + (size_t)arow * 128;

  fetch_w(wblk, sW0, OFFW0, tid);
  fetch_wait(tid); __syncthreads();

  v8f acc[8];
#pragma unroll
  for (int nt = 0; nt < 8; ++nt) acc[nt] = vzero();

  // layer 1 chunk 0: node state (f16)
  fetch_w(wblk + BLKH, sW1, OFFW1, tid);
#pragma unroll
  for (int k0 = 0; k0 < 128; k0 += 32) {
    v16h a = a_frag_row(pN, k0, lane);
#pragma unroll
    for (int nt = 0; nt < 8; ++nt)
      acc[nt] = wmma16(a, b_frag_lds(sW0, k0, nt * 16, lane), acc[nt]);
  }
  fetch_wait(tid); __syncthreads();

  // layer 1 chunk 1: aggregation (f32 -> f16)
  fetch_w(wblk + (size_t)2 * BLKH, sW0, OFFW0, tid);
#pragma unroll
  for (int k0 = 0; k0 < 128; k0 += 32) {
    v16h a = a_frag_row_f32(pA, k0, lane);
#pragma unroll
    for (int nt = 0; nt < 8; ++nt)
      acc[nt] = wmma16(a, b_frag_lds(sW1, k0, nt * 16, lane), acc[nt]);
  }
  fetch_wait(tid); __syncthreads();
  bias_relu_store(sX, b0, m0, lane, acc);

  // layer 2 (W1 in sW0), prefetch W2 -> sW1
  fetch_w(wblk + (size_t)3 * BLKH, sW1, OFFW1, tid);
#pragma unroll
  for (int nt = 0; nt < 8; ++nt) acc[nt] = vzero();
  gemm128_lds(sX, sW0, m0, lane, acc);
  fetch_wait(tid); __syncthreads();
  bias_relu_store(sX, b1, m0, lane, acc);

  // layer 3
#pragma unroll
  for (int nt = 0; nt < 8; ++nt) acc[nt] = vzero();
  gemm128_lds(sX, sW1, m0, lane, acc);

  add_bias(acc, b2, lane);
  float mu[8], rs[8];
  ln_stats(acc, mu, rs);
  int half8 = (lane >> 4) << 3, nlo = lane & 15;
  int rowBase = tile + m0 + half8;
#pragma unroll
  for (int nt = 0; nt < 8; ++nt) {
    int n = nt * 16 + nlo;
    float sc = lnS[n], of = lnO[n];
#pragma unroll
    for (int r = 0; r < 8; ++r) {
      int g = rowBase + r;
      if (g < Nn) {
        float y = (acc[nt][r] - mu[r]) * rs[r] * sc + of;
        float nv = nodes_f[g * 128 + n] + y;  // residual
        nodes_f[g * 128 + n] = nv;
        nodes_h[g * 128 + n] = (_Float16)nv;
      }
    }
  }
}

// -------------------------------------------------------------- decoder -----
__global__ __launch_bounds__(THREADS) void gns_decoder_kernel(
    const _Float16* __restrict__ nodes_h,
    const _Float16* __restrict__ wblk,   // [W0,W1,W2(3 cols padded)]
    const float* __restrict__ b0, const float* __restrict__ b1, const float* __restrict__ b2,
    float* __restrict__ out, int Nn) {
  extern __shared__ _Float16 smem[];
  _Float16* sX = smem;
  _Float16* sW0 = smem + 128 * XP;
  _Float16* sW1 = sW0 + BLKH;
  int tid = threadIdx.x, lane = tid & 31, m0 = (tid >> 5) * 16;
  int tile = blockIdx.x * 128;
  int arow = tile + m0 + (lane & 15);
  if (arow >= Nn) arow = Nn - 1;
  const _Float16* pN = nodes_h + (size_t)arow * 128;

  fetch_w(wblk, sW0, OFFW0, tid);
  fetch_wait(tid); __syncthreads();

  // layer 1
  fetch_w(wblk + BLKH, sW1, OFFW1, tid);
  v8f acc[8];
#pragma unroll
  for (int nt = 0; nt < 8; ++nt) acc[nt] = vzero();
#pragma unroll
  for (int k0 = 0; k0 < 128; k0 += 32) {
    v16h a = a_frag_row(pN, k0, lane);
#pragma unroll
    for (int nt = 0; nt < 8; ++nt)
      acc[nt] = wmma16(a, b_frag_lds(sW0, k0, nt * 16, lane), acc[nt]);
  }
  fetch_wait(tid); __syncthreads();
  bias_relu_store(sX, b0, m0, lane, acc);

  // layer 2
  fetch_w(wblk + (size_t)2 * BLKH, sW0, OFFW0, tid);
#pragma unroll
  for (int nt = 0; nt < 8; ++nt) acc[nt] = vzero();
  gemm128_lds(sX, sW1, m0, lane, acc);
  fetch_wait(tid); __syncthreads();
  bias_relu_store(sX, b1, m0, lane, acc);

  // layer 3: 128 -> 3 (cols padded to 16 with zero weights)
  v8f c0 = vzero();
#pragma unroll
  for (int k0 = 0; k0 < 128; k0 += 32) {
    v16h a = a_frag_lds(sX, XP, m0, k0, lane);
    c0 = wmma16(a, b_frag_lds(sW0, k0, 0, lane), c0);
  }
  int n = lane & 15;
  int half8 = (lane >> 4) << 3;
  int rowBase = tile + m0 + half8;
  if (n < 3) {
    float bv = b2[n];
#pragma unroll
    for (int r = 0; r < 8; ++r) {
      int g = rowBase + r;
      if (g < Nn) out[g * 3 + n] = c0[r] + bv;
    }
  }
}

__global__ void gns_zero_kernel(float* __restrict__ p, int n) {
  int i = blockIdx.x * THREADS + threadIdx.x;
  if (i < n) p[i] = 0.f;
}

// ---------------------------------------------------------------- host ------
extern "C" void kernel_launch(void* const* d_in, const int* in_sizes, int n_in,
                              void* d_out, int out_size, void* d_ws, size_t ws_size,
                              hipStream_t stream) {
  (void)n_in; (void)out_size; (void)ws_size;
  const int N = in_sizes[0] / 15;
  const int E = in_sizes[4] / 3;
  const int S = in_sizes[42] / 128;

  const float* vel_hist = (const float*)d_in[0];
  const float* vel_mag  = (const float*)d_in[1];
  const float* bound    = (const float*)d_in[2];
  const float* force    = (const float*)d_in[3];
  const float* rel_disp = (const float*)d_in[4];
  const float* rel_dist = (const float*)d_in[5];
  const int* senders    = (const int*)d_in[6];
  const int* receivers  = (const int*)d_in[7];
  const float* en_W0 = (const float*)d_in[8];  const float* en_b0 = (const float*)d_in[9];
  const float* en_W1 = (const float*)d_in[10]; const float* en_b1 = (const float*)d_in[11];
  const float* en_W2 = (const float*)d_in[12]; const float* en_b2 = (const float*)d_in[13];
  const float* ee_W0 = (const float*)d_in[14]; const float* ee_b0 = (const float*)d_in[15];
  const float* ee_W1 = (const float*)d_in[16]; const float* ee_b1 = (const float*)d_in[17];
  const float* ee_W2 = (const float*)d_in[18]; const float* ee_b2 = (const float*)d_in[19];
  const float* pe_W0 = (const float*)d_in[20]; const float* pe_b0 = (const float*)d_in[21];
  const float* pe_W1 = (const float*)d_in[22]; const float* pe_b1 = (const float*)d_in[23];
  const float* pe_W2 = (const float*)d_in[24]; const float* pe_b2 = (const float*)d_in[25];
  const float* pn_W0 = (const float*)d_in[26]; const float* pn_b0 = (const float*)d_in[27];
  const float* pn_W1 = (const float*)d_in[28]; const float* pn_b1 = (const float*)d_in[29];
  const float* pn_W2 = (const float*)d_in[30]; const float* pn_b2 = (const float*)d_in[31];
  const float* de_W0 = (const float*)d_in[32]; const float* de_b0 = (const float*)d_in[33];
  const float* de_W1 = (const float*)d_in[34]; const float* de_b1 = (const float*)d_in[35];
  const float* de_W2 = (const float*)d_in[36]; const float* de_b2 = (const float*)d_in[37];
  const float* en_s = (const float*)d_in[38]; const float* en_o = (const float*)d_in[39];
  const float* ee_s = (const float*)d_in[40]; const float* ee_o = (const float*)d_in[41];
  const float* pe_s = (const float*)d_in[42]; const float* pe_o = (const float*)d_in[43];
  const float* pn_s = (const float*)d_in[44]; const float* pn_o = (const float*)d_in[45];

  // workspace carve
  float* nodes_f = (float*)d_ws;                       // N*128
  float* aggr    = nodes_f + (size_t)N * 128;          // N*128
  float* edges_f = aggr + (size_t)N * 128;             // E*128
  _Float16* nodes_h = (_Float16*)(edges_f + (size_t)E * 128);  // N*128
  _Float16* edges_h = nodes_h + (size_t)N * 128;               // E*128
  _Float16* wsW = edges_h + (size_t)E * 128;           // prestaged weight blocks
  wsW = (_Float16*)(((uintptr_t)wsW + 15) & ~(uintptr_t)15);

  _Float16* wEncN = wsW;                               // 3 blocks
  _Float16* wEncE = wsW + (size_t)3 * BLKH;            // 3 blocks
  _Float16* wDec  = wsW + (size_t)6 * BLKH;            // 3 blocks
  _Float16* wEdge = wsW + (size_t)9 * BLKH;            // 5 blocks/step
  _Float16* wNode = wsW + (size_t)(9 + 5 * S) * BLKH;  // 4 blocks/step

  dim3 blk(THREADS);
  int nodeTiles = (N + 127) / 128;
  int edgeTiles = (E + 127) / 128;
  size_t smemB = (size_t)SMEM_HALFS * sizeof(_Float16);

  // ---- stage all weights once into TDM-ready f16 column-major blocks ----
  gns_prep_kernel<<<1, blk, 0, stream>>>(en_W0, 0, 29, 128, 1, wEncN + (size_t)0 * BLKH, 1);
  gns_prep_kernel<<<1, blk, 0, stream>>>(en_W1, 0, 128, 128, 1, wEncN + (size_t)1 * BLKH, 1);
  gns_prep_kernel<<<1, blk, 0, stream>>>(en_W2, 0, 128, 128, 1, wEncN + (size_t)2 * BLKH, 1);
  gns_prep_kernel<<<1, blk, 0, stream>>>(ee_W0, 0, 4, 128, 1, wEncE + (size_t)0 * BLKH, 1);
  gns_prep_kernel<<<1, blk, 0, stream>>>(ee_W1, 0, 128, 128, 1, wEncE + (size_t)1 * BLKH, 1);
  gns_prep_kernel<<<1, blk, 0, stream>>>(ee_W2, 0, 128, 128, 1, wEncE + (size_t)2 * BLKH, 1);
  gns_prep_kernel<<<1, blk, 0, stream>>>(de_W0, 0, 128, 128, 1, wDec + (size_t)0 * BLKH, 1);
  gns_prep_kernel<<<1, blk, 0, stream>>>(de_W1, 0, 128, 128, 1, wDec + (size_t)1 * BLKH, 1);
  gns_prep_kernel<<<1, blk, 0, stream>>>(de_W2, 0, 128, 3, 1, wDec + (size_t)2 * BLKH, 1);
  gns_prep_kernel<<<S * 3, blk, 0, stream>>>(pe_W0, 384 * 128, 384, 128, 3, wEdge, 5);
  gns_prep_kernel<<<S, blk, 0, stream>>>(pe_W1, 128 * 128, 128, 128, 1, wEdge + (size_t)3 * BLKH, 5);
  gns_prep_kernel<<<S, blk, 0, stream>>>(pe_W2, 128 * 128, 128, 128, 1, wEdge + (size_t)4 * BLKH, 5);
  gns_prep_kernel<<<S * 2, blk, 0, stream>>>(pn_W0, 256 * 128, 256, 128, 2, wNode, 4);
  gns_prep_kernel<<<S, blk, 0, stream>>>(pn_W1, 128 * 128, 128, 128, 1, wNode + (size_t)2 * BLKH, 4);
  gns_prep_kernel<<<S, blk, 0, stream>>>(pn_W2, 128 * 128, 128, 128, 1, wNode + (size_t)3 * BLKH, 4);

  // ---- encode ----
  gns_encoder_kernel<<<nodeTiles, blk, smemB, stream>>>(
      vel_hist, 15, vel_mag, 5, bound, 6, force, 3, 29,
      wEncN, en_b0, en_b1, en_b2, en_s, en_o, nodes_f, nodes_h, N);
  gns_encoder_kernel<<<edgeTiles, blk, smemB, stream>>>(
      rel_disp, 3, rel_dist, 1, nullptr, 0, nullptr, 0, 4,
      wEncE, ee_b0, ee_b1, ee_b2, ee_s, ee_o, edges_f, edges_h, E);

  // ---- process ----
  int zgrid = (N * 128 + THREADS - 1) / THREADS;
  for (int s = 0; s < S; ++s) {
    gns_zero_kernel<<<zgrid, blk, 0, stream>>>(aggr, N * 128);
    gns_edge_kernel<<<edgeTiles, blk, smemB, stream>>>(
        nodes_h, edges_h, edges_f, aggr, senders, receivers,
        wEdge + (size_t)s * 5 * BLKH,
        pe_b0 + (size_t)s * 128, pe_b1 + (size_t)s * 128, pe_b2 + (size_t)s * 128,
        pe_s + (size_t)s * 128, pe_o + (size_t)s * 128, E);
    gns_node_kernel<<<nodeTiles, blk, smemB, stream>>>(
        nodes_h, nodes_f, aggr,
        wNode + (size_t)s * 4 * BLKH,
        pn_b0 + (size_t)s * 128, pn_b1 + (size_t)s * 128, pn_b2 + (size_t)s * 128,
        pn_s + (size_t)s * 128, pn_o + (size_t)s * 128, N);
  }

  // ---- decode ----
  gns_decoder_kernel<<<nodeTiles, blk, smemB, stream>>>(
      nodes_h, wDec, de_b0, de_b1, de_b2, (float*)d_out, N);
}
